// Model_46952582480549
// MI455X (gfx1250) — matrix operand order repaired and checked by reading, not compiled, not run on hardware
//
#include <hip/hip_runtime.h>
#include <math.h>

#define NN 50000
#define NE 640000
#define DD 128
#define LL 4
#define FN 64
#define FE 32
#define BN_EPS 1e-5f

typedef __bf16 bf16;
typedef bf16 v16bf __attribute__((ext_vector_type(16)));
typedef float v8f  __attribute__((ext_vector_type(8)));

// ---- order-preserving float<->uint for atomic max ----
__device__ __forceinline__ unsigned fflip(float f) {
    unsigned u = __float_as_uint(f);
    return (u & 0x80000000u) ? ~u : (u | 0x80000000u);
}
__device__ __forceinline__ float funflip(unsigned u) {
    return (u & 0x80000000u) ? __uint_as_float(u & 0x7FFFFFFFu)
                             : __uint_as_float(~u);
}
#define FLIP_NEG_INF 0x007FFFFFu  // fflip(-inf)

__device__ __forceinline__ float sigm(float x) { return 1.0f / (1.0f + expf(-x)); }

// =====================================================================
// Generic WMMA GEMM: C[M x 128] (+)= gather(A)[M x K] * W[K x 128] (+ bias)
// block = 128 threads (4 wave32), each wave computes a 16 x 128 strip.
// K % 32 == 0, M % 16 == 0 (true for all call sites here).
// =====================================================================
template<bool GATHER, bool BIAS, bool ACCUM>
__global__ void __launch_bounds__(128)
gemm_wmma(const float* __restrict__ A, const int* __restrict__ gidx,
          int M, int K, int lda,
          const float* __restrict__ W, const float* __restrict__ bias,
          float* __restrict__ C)
{
    __shared__ bf16 lds_w[128 * 34];   // W tile transposed: [n][k], pad 34
    const int tid  = threadIdx.x;
    const int lane = tid & 31;
    const int wave = tid >> 5;
    const int mrow = lane & 15;
    const int half = lane >> 4;
    const int rowTileBase = blockIdx.x * 64 + wave * 16;
    const bool active = rowTileBase < M;

    long rowIn = 0;
    if (active) {
        const int rOut = rowTileBase + mrow;
        rowIn = GATHER ? (long)gidx[rOut] : (long)rOut;
    }
    const float* arow0 = A + rowIn * (long)lda;

    v8f acc[8] = {};

    for (int k0 = 0; k0 < K; k0 += 32) {
        // stage 32 x 128 fp32 weights -> bf16 transposed LDS (thread = column)
        #pragma unroll 4
        for (int kk = 0; kk < 32; ++kk)
            lds_w[tid * 34 + kk] = (bf16)W[(size_t)(k0 + kk) * 128 + tid];
        __syncthreads();

        if (active) {
            const float* ar = arow0 + k0;
            // A fragment: 16x32 bf16 layout (lanes 0-15: K 0..7 / 16..23,
            //             lanes 16-31: K 8..15 / 24..31)
            v16bf a;
            #pragma unroll
            for (int j = 0; j < 8; ++j)  a[j] = (bf16)ar[j + 8 * half];
            #pragma unroll
            for (int j = 8; j < 16; ++j) a[j] = (bf16)ar[j + 8 + 8 * half];

            #pragma unroll
            for (int nt = 0; nt < 8; ++nt) {
                const bf16* wc = &lds_w[(nt * 16 + mrow) * 34 + 8 * half];
                v16bf b;
                #pragma unroll
                for (int j = 0; j < 16; ++j) b[j] = wc[(j & 7) + ((j >> 3) << 4)];
                acc[nt] = __builtin_amdgcn_wmma_f32_16x16x32_bf16(
                    false, a, false, b, (short)0, acc[nt], false, false);
            }
        }
        __syncthreads();
    }

    if (active) {
        #pragma unroll
        for (int nt = 0; nt < 8; ++nt) {
            const int col = nt * 16 + mrow;
            const float bcol = BIAS ? bias[col] : 0.0f;
            #pragma unroll
            for (int r = 0; r < 8; ++r) {
                const long row = rowTileBase + r + 8 * half;
                float v = acc[nt][r] + bcol;
                float* p = C + row * 128 + col;
                if (ACCUM) v += *p;
                *p = v;
            }
        }
    }
}

// =====================================================================
// Edge fused pass: tmp_e = eA + hB[dst] + hC[src]  (eA pre-stored in tmp_e)
//                  msg   = hV[src] * sigmoid(e) -> atomicMax into agg[dst]
//                  BN stats (sum, sumsq) for tmp_e
// thread = (edge, 4 consecutive features); stride keeps column fixed.
// =====================================================================
__global__ void __launch_bounds__(256)
edge_pre(float* __restrict__ tmp_e, const float* __restrict__ e,
         const float* __restrict__ hV, const float* __restrict__ hB,
         const float* __restrict__ hC, const int* __restrict__ src,
         const int* __restrict__ dst, unsigned* __restrict__ agg,
         float* __restrict__ sum_e, float* __restrict__ sq_e)
{
    __shared__ float ssum[128], ssq[128];
    const long tid    = (long)blockIdx.x * blockDim.x + threadIdx.x;
    const long stride = (long)gridDim.x * blockDim.x;    // multiple of 32
    const int  col    = ((int)(tid & 31)) * 4;
    float s0=0,s1=0,s2=0,s3=0, q0=0,q1=0,q2=0,q3=0;
    const long total = (long)NE * 32;

    for (long idx = tid; idx < total; idx += stride) {
        const int i  = (int)(idx >> 5);
        const int sN = src[i], dN = dst[i];
        const float4 ea = *(const float4*)(tmp_e + (long)i  * 128 + col);
        const float4 hb = *(const float4*)(hB    + (long)dN * 128 + col);
        const float4 hc = *(const float4*)(hC    + (long)sN * 128 + col);
        float4 t;
        t.x = ea.x + hb.x + hc.x;  t.y = ea.y + hb.y + hc.y;
        t.z = ea.z + hb.z + hc.z;  t.w = ea.w + hb.w + hc.w;
        *(float4*)(tmp_e + (long)i * 128 + col) = t;
        s0 += t.x; s1 += t.y; s2 += t.z; s3 += t.w;
        q0 += t.x*t.x; q1 += t.y*t.y; q2 += t.z*t.z; q3 += t.w*t.w;

        const float4 ev = *(const float4*)(e  + (long)i  * 128 + col);
        const float4 hv = *(const float4*)(hV + (long)sN * 128 + col);
        unsigned* ap = agg + (long)dN * 128 + col;
        atomicMax(ap + 0, fflip(hv.x * sigm(ev.x)));
        atomicMax(ap + 1, fflip(hv.y * sigm(ev.y)));
        atomicMax(ap + 2, fflip(hv.z * sigm(ev.z)));
        atomicMax(ap + 3, fflip(hv.w * sigm(ev.w)));
    }

    if (threadIdx.x < 128) { ssum[threadIdx.x] = 0.f; ssq[threadIdx.x] = 0.f; }
    __syncthreads();
    atomicAdd(&ssum[col+0], s0); atomicAdd(&ssum[col+1], s1);
    atomicAdd(&ssum[col+2], s2); atomicAdd(&ssum[col+3], s3);
    atomicAdd(&ssq[col+0], q0);  atomicAdd(&ssq[col+1], q1);
    atomicAdd(&ssq[col+2], q2);  atomicAdd(&ssq[col+3], q3);
    __syncthreads();
    if (threadIdx.x < 128) {
        atomicAdd(&sum_e[threadIdx.x], ssum[threadIdx.x]);
        atomicAdd(&sq_e[threadIdx.x],  ssq[threadIdx.x]);
    }
}

// tmp_n = hU + decode(agg)  (0 fill for 0-in-degree) + node BN stats
__global__ void __launch_bounds__(256)
node_pre(float* __restrict__ tmp_n, const unsigned* __restrict__ agg,
         float* __restrict__ sum_n, float* __restrict__ sq_n)
{
    __shared__ float ssum[128], ssq[128];
    const long tid    = (long)blockIdx.x * blockDim.x + threadIdx.x;
    const long stride = (long)gridDim.x * blockDim.x;
    const int  col    = ((int)(tid & 31)) * 4;
    float s0=0,s1=0,s2=0,s3=0, q0=0,q1=0,q2=0,q3=0;
    const long total = (long)NN * 32;

    for (long idx = tid; idx < total; idx += stride) {
        const long base = (idx >> 5) * 128 + col;
        float4 t = *(const float4*)(tmp_n + base);
        float a0 = funflip(agg[base+0]); if (a0 == -INFINITY) a0 = 0.f;
        float a1 = funflip(agg[base+1]); if (a1 == -INFINITY) a1 = 0.f;
        float a2 = funflip(agg[base+2]); if (a2 == -INFINITY) a2 = 0.f;
        float a3 = funflip(agg[base+3]); if (a3 == -INFINITY) a3 = 0.f;
        t.x += a0; t.y += a1; t.z += a2; t.w += a3;
        *(float4*)(tmp_n + base) = t;
        s0 += t.x; s1 += t.y; s2 += t.z; s3 += t.w;
        q0 += t.x*t.x; q1 += t.y*t.y; q2 += t.z*t.z; q3 += t.w*t.w;
    }
    if (threadIdx.x < 128) { ssum[threadIdx.x] = 0.f; ssq[threadIdx.x] = 0.f; }
    __syncthreads();
    atomicAdd(&ssum[col+0], s0); atomicAdd(&ssum[col+1], s1);
    atomicAdd(&ssum[col+2], s2); atomicAdd(&ssum[col+3], s3);
    atomicAdd(&ssq[col+0], q0);  atomicAdd(&ssq[col+1], q1);
    atomicAdd(&ssq[col+2], q2);  atomicAdd(&ssq[col+3], q3);
    __syncthreads();
    if (threadIdx.x < 128) {
        atomicAdd(&sum_n[threadIdx.x], ssum[threadIdx.x]);
        atomicAdd(&sq_n[threadIdx.x],  ssq[threadIdx.x]);
    }
}

// sums[0:128]=sum_n, [128:256]=sumsq_n, [256:384]=sum_e, [384:512]=sumsq_e
// out: mean_n, istd_n, mean_e, istd_e
__global__ void finalize_stats(const float* __restrict__ sums, float* __restrict__ out)
{
    const int f = threadIdx.x;
    if (f < 128) {
        const float mn = sums[f] * (1.0f / NN);
        const float vn = sums[128+f] * (1.0f / NN) - mn * mn;
        out[f]       = mn;
        out[128 + f] = rsqrtf(vn + BN_EPS);
        const float me = sums[256+f] * (1.0f / NE);
        const float ve = sums[384+f] * (1.0f / NE) - me * me;
        out[256 + f] = me;
        out[384 + f] = rsqrtf(ve + BN_EPS);
    }
}

// out = x + relu((t - mean) * istd)
__global__ void __launch_bounds__(256)
bn_apply(const float* __restrict__ x, const float* __restrict__ t,
         const float* __restrict__ mean, const float* __restrict__ istd,
         float* __restrict__ out, long M)
{
    const long tid    = (long)blockIdx.x * blockDim.x + threadIdx.x;
    const long stride = (long)gridDim.x * blockDim.x;
    const int  col    = ((int)(tid & 31)) * 4;
    const float m0 = mean[col+0], m1 = mean[col+1], m2 = mean[col+2], m3 = mean[col+3];
    const float i0 = istd[col+0], i1 = istd[col+1], i2 = istd[col+2], i3 = istd[col+3];
    const long total = M * 32;
    for (long idx = tid; idx < total; idx += stride) {
        const long base = (idx >> 5) * 128 + col;
        const float4 tv = *(const float4*)(t + base);
        const float4 xv = *(const float4*)(x + base);
        float4 o;
        o.x = xv.x + fmaxf((tv.x - m0) * i0, 0.f);
        o.y = xv.y + fmaxf((tv.y - m1) * i1, 0.f);
        o.z = xv.z + fmaxf((tv.z - m2) * i2, 0.f);
        o.w = xv.w + fmaxf((tv.w - m3) * i3, 0.f);
        *(float4*)(out + base) = o;
    }
}

// per-feature column sums of x[M x 128] into out[128] (prezeroed)
__global__ void __launch_bounds__(256)
colsum(const float* __restrict__ x, long M, float* __restrict__ out)
{
    __shared__ float ssum[128];
    const long tid    = (long)blockIdx.x * blockDim.x + threadIdx.x;
    const long stride = (long)gridDim.x * blockDim.x;
    const int  col    = ((int)(tid & 31)) * 4;
    float s0=0,s1=0,s2=0,s3=0;
    const long total = M * 32;
    for (long idx = tid; idx < total; idx += stride) {
        const float4 v = *(const float4*)(x + (idx >> 5) * 128 + col);
        s0 += v.x; s1 += v.y; s2 += v.z; s3 += v.w;
    }
    if (threadIdx.x < 128) ssum[threadIdx.x] = 0.f;
    __syncthreads();
    atomicAdd(&ssum[col+0], s0); atomicAdd(&ssum[col+1], s1);
    atomicAdd(&ssum[col+2], s2); atomicAdd(&ssum[col+3], s3);
    __syncthreads();
    if (threadIdx.x < 128) atomicAdd(&out[threadIdx.x], ssum[threadIdx.x]);
}

// c0[n] = b1[n] + sum_k (hsum[k]/N) * W1_top[k][n]
__global__ void c0_kernel(const float* __restrict__ hsum, const float* __restrict__ W1top,
                          const float* __restrict__ b1, float* __restrict__ c0)
{
    __shared__ float s[128];
    const int n = threadIdx.x;
    if (n < 128) s[n] = hsum[n] * (1.0f / NN);
    __syncthreads();
    if (n < 128) {
        float c = b1[n];
        for (int k = 0; k < 128; ++k) c += s[k] * W1top[k * 128 + n];
        c0[n] = c;
    }
}

// out[i] = sigmoid(b2 + sum_c relu(z[i,c]) * W2[c]); one wave32 per edge
__global__ void __launch_bounds__(256)
final_out(const float* __restrict__ z, const float* __restrict__ W2,
          const float* __restrict__ b2, float* __restrict__ out)
{
    const int edge = blockIdx.x * 8 + (threadIdx.x >> 5);
    const int lane = threadIdx.x & 31;
    if (edge >= NE) return;
    float acc = 0.f;
    #pragma unroll
    for (int j = 0; j < 4; ++j) {
        const int c = lane + j * 32;
        acc += fmaxf(z[(long)edge * 128 + c], 0.f) * W2[c];
    }
    #pragma unroll
    for (int off = 16; off > 0; off >>= 1) acc += __shfl_xor(acc, off, 32);
    if (lane == 0) out[edge] = 1.0f / (1.0f + expf(-(acc + b2[0])));
}

__global__ void fill_f32(float* p, float v, long n) {
    const long tid = (long)blockIdx.x * blockDim.x + threadIdx.x;
    const long st  = (long)gridDim.x * blockDim.x;
    for (long i = tid; i < n; i += st) p[i] = v;
}
__global__ void fill_u32(unsigned* p, unsigned v, long n) {
    const long tid = (long)blockIdx.x * blockDim.x + threadIdx.x;
    const long st  = (long)gridDim.x * blockDim.x;
    for (long i = tid; i < n; i += st) p[i] = v;
}

extern "C" void kernel_launch(void* const* d_in, const int* in_sizes, int n_in,
                              void* d_out, int out_size, void* d_ws, size_t ws_size,
                              hipStream_t stream)
{
    const float* h_in    = (const float*)d_in[0];
    const float* e_in    = (const float*)d_in[1];
    const int*   src     = (const int*)d_in[2];
    const int*   dst     = (const int*)d_in[3];
    const float* w_emb_n = (const float*)d_in[4];
    const float* b_emb_n = (const float*)d_in[5];
    const float* w_emb_e = (const float*)d_in[6];
    const float* b_emb_e = (const float*)d_in[7];
    const float* Umat    = (const float*)d_in[8];
    const float* Vmat    = (const float*)d_in[9];
    const float* Amat    = (const float*)d_in[10];
    const float* Bmat    = (const float*)d_in[11];
    const float* Cmat    = (const float*)d_in[12];
    const float* W1      = (const float*)d_in[13];
    const float* b1      = (const float*)d_in[14];
    const float* W2      = (const float*)d_in[15];
    const float* b2      = (const float*)d_in[16];
    float* out = (float*)d_out;

    float* ws = (float*)d_ws;
    const size_t ND = (size_t)NN * DD, ED = (size_t)NE * DD;
    float* h0    = ws; ws += ND;
    float* h1    = ws; ws += ND;
    float* e0    = ws; ws += ED;
    float* tmpe  = ws; ws += ED;   // also eA scratch + MLP hidden z
    float* hV    = ws; ws += ND;
    float* hB    = ws; ws += ND;
    float* hC    = ws; ws += ND;
    float* tmpn  = ws; ws += ND;   // holds hU, then hU + agg
    unsigned* agg = (unsigned*)ws; ws += ND;
    float* ssum  = ws; ws += 512;  // sum_n, sq_n, sum_e, sq_e
    float* sout  = ws; ws += 512;  // mean_n, istd_n, mean_e, istd_e
    float* hsum  = ws; ws += 128;
    float* c0    = ws; ws += 128;

    const dim3 B128(128), B256(256);
    const int gN = (NN + 63) / 64, gE = (NE + 63) / 64;

    // embeddings
    gemm_wmma<false, true, false><<<gN, B128, 0, stream>>>(h_in, nullptr, NN, FN, FN, w_emb_n, b_emb_n, h0);
    gemm_wmma<false, true, false><<<gE, B128, 0, stream>>>(e_in, nullptr, NE, FE, FE, w_emb_e, b_emb_e, e0);

    float* h = h0; float* hn = h1;
    for (int l = 0; l < LL; ++l) {
        const float* Ul = Umat + (size_t)l * DD * DD;
        const float* Vl = Vmat + (size_t)l * DD * DD;
        const float* Al = Amat + (size_t)l * DD * DD;
        const float* Bl = Bmat + (size_t)l * DD * DD;
        const float* Cl = Cmat + (size_t)l * DD * DD;

        fill_u32<<<1024, B256, 0, stream>>>(agg, FLIP_NEG_INF, (long)ND);
        fill_f32<<<2, B256, 0, stream>>>(ssum, 0.f, 512);

        gemm_wmma<false, false, false><<<gN, B128, 0, stream>>>(h, nullptr, NN, DD, DD, Vl, nullptr, hV);
        gemm_wmma<false, false, false><<<gN, B128, 0, stream>>>(h, nullptr, NN, DD, DD, Ul, nullptr, tmpn);
        gemm_wmma<false, false, false><<<gN, B128, 0, stream>>>(h, nullptr, NN, DD, DD, Bl, nullptr, hB);
        gemm_wmma<false, false, false><<<gN, B128, 0, stream>>>(h, nullptr, NN, DD, DD, Cl, nullptr, hC);
        gemm_wmma<false, false, false><<<gE, B128, 0, stream>>>(e0, nullptr, NE, DD, DD, Al, nullptr, tmpe);

        edge_pre<<<2048, B256, 0, stream>>>(tmpe, e0, hV, hB, hC, src, dst, agg, ssum + 256, ssum + 384);
        node_pre<<<512, B256, 0, stream>>>(tmpn, agg, ssum, ssum + 128);
        finalize_stats<<<1, B128, 0, stream>>>(ssum, sout);

        bn_apply<<<512, B256, 0, stream>>>(h, tmpn, sout, sout + 128, hn, (long)NN);
        bn_apply<<<2048, B256, 0, stream>>>(e0, tmpe, sout + 256, sout + 384, e0, (long)NE);

        float* t = h; h = hn; hn = t;
    }

    // readout
    fill_f32<<<1, B256, 0, stream>>>(hsum, 0.f, 128);
    colsum<<<512, B256, 0, stream>>>(h, (long)NN, hsum);
    c0_kernel<<<1, B128, 0, stream>>>(hsum, W1, b1, c0);

    gemm_wmma<true, false, false><<<gE, B128, 0, stream>>>(h, src, NE, DD, DD, W1 + 128 * 128, nullptr, tmpe);
    gemm_wmma<true, true,  true ><<<gE, B128, 0, stream>>>(h, dst, NE, DD, DD, W1 + 256 * 128, c0, tmpe);

    final_out<<<(NE + 7) / 8, B256, 0, stream>>>(tmpe, W2, b2, out);
}
